// SecondStageDetector_79989470920813
// MI455X (gfx1250) — compile-verified, matrix operand order checked by reading, not run on hardware
//
#include <hip/hip_runtime.h>
#include <hip/hip_bf16.h>
#include <math.h>

#define ROI    7
#define NBOX   1024
#define CH     32
#define D_IN   1568   // 7*7*32
#define HD     2048
#define NMS_THRV 0.01f
#define NMS_OUT  100

typedef __attribute__((ext_vector_type(16))) __bf16 v16bf;
typedef __attribute__((ext_vector_type(8)))  float  v8f;
typedef int v4i_vs __attribute__((vector_size(16)));   // matches builtin param type

struct alignas(16) U128 { unsigned int w[4]; };

#if defined(__has_builtin)
#  if __has_builtin(__builtin_amdgcn_global_load_async_to_lds_b128)
#    define HAVE_ASYNC_LDS 1
#  endif
#endif

// Async global->LDS copy (ASYNCcnt path on gfx1250) with sync fallback.
__device__ __forceinline__ void async_copy_b128(void* lds, const void* g) {
#ifdef HAVE_ASYNC_LDS
    __builtin_amdgcn_global_load_async_to_lds_b128(
        (__attribute__((address_space(1))) v4i_vs*)(g),
        (__attribute__((address_space(3))) v4i_vs*)(lds), 0, 0);
#else
    *(U128*)lds = *(const U128*)g;
#endif
}

__device__ __forceinline__ void wait_async_copies() {
#ifdef HAVE_ASYNC_LDS
#  if __has_builtin(__builtin_amdgcn_s_wait_asynccnt)
    __builtin_amdgcn_s_wait_asynccnt(0);
#  else
    asm volatile("s_wait_asynccnt 0x0" ::: "memory");
#  endif
#endif
}

__device__ __forceinline__ __bf16 f2bf(float f) {
    unsigned u = __builtin_bit_cast(unsigned, f);
    unsigned r = (u + 0x7FFFu + ((u >> 16) & 1u)) >> 16;
    unsigned short s = (unsigned short)r;
    return __builtin_bit_cast(__bf16, s);
}

// ---------------------------------------------------------------- ROI align
__device__ __forceinline__ float bilinear(const float* __restrict__ f, int H, int W,
                                          float yn, float xn, int c) {
    float ys = yn * (float)(H - 1);
    float xs = xn * (float)(W - 1);
    float yf = floorf(ys), xf = floorf(xs);
    int y0 = (int)fminf(fmaxf(yf, 0.f), (float)(H - 1));
    int x0 = (int)fminf(fmaxf(xf, 0.f), (float)(W - 1));
    int y1 = (y0 + 1 < H) ? y0 + 1 : H - 1;
    int x1 = (x0 + 1 < W) ? x0 + 1 : W - 1;
    float wy = ys - (float)y0;
    float wx = xs - (float)x0;
    float f00 = f[((long long)y0 * W + x0) * CH + c];
    float f01 = f[((long long)y0 * W + x1) * CH + c];
    float f10 = f[((long long)y1 * W + x0) * CH + c];
    float f11 = f[((long long)y1 * W + x1) * CH + c];
    return f00 * (1.f - wy) * (1.f - wx) + f01 * (1.f - wy) * wx
         + f10 * wy * (1.f - wx)        + f11 * wy * wx;
}

__global__ void __launch_bounds__(256)
roi_fuse_kernel(const float* __restrict__ imgF, const float* __restrict__ bevF,
                const float* __restrict__ imgB, const float* __restrict__ bevB,
                const float* __restrict__ imgM, const float* __restrict__ bevM,
                __bf16* __restrict__ X) {
    int b = blockIdx.x;
    float mi = imgM[0], mb = bevM[0];
    float inv = 1.0f / (mi + mb);
    float iy0 = imgB[b * 4 + 0], ix0 = imgB[b * 4 + 1];
    float iy1 = imgB[b * 4 + 2], ix1 = imgB[b * 4 + 3];
    float by0 = bevB[b * 4 + 0], bx0 = bevB[b * 4 + 1];
    float by1 = bevB[b * 4 + 2], bx1 = bevB[b * 4 + 3];
    for (int e = threadIdx.x; e < D_IN; e += blockDim.x) {
        int c = e & 31;
        int j = (e >> 5) % ROI;
        int i = e / (ROI * CH);
        float ti = (float)i * (1.0f / (ROI - 1));
        float tj = (float)j * (1.0f / (ROI - 1));
        float vi = bilinear(imgF, 360, 1200, iy0 + (iy1 - iy0) * ti, ix0 + (ix1 - ix0) * tj, c);
        float vb = bilinear(bevF, 700,  800, by0 + (by1 - by0) * ti, bx0 + (bx1 - bx0) * tj, c);
        float fused = (mi * vi + mb * vb) * inv;
        X[(long long)b * D_IN + e] = f2bf(fused);
    }
}

// ----------------------------------------------- weight transpose + bf16 cvt
__global__ void __launch_bounds__(256)
wconv_kernel(const float* __restrict__ W, __bf16* __restrict__ Wt, int K, int N) {
    long long idx = (long long)blockIdx.x * blockDim.x + threadIdx.x;
    long long total = (long long)K * N;
    if (idx >= total) return;
    int k = (int)(idx / N);
    int n = (int)(idx % N);
    Wt[(long long)n * K + k] = f2bf(W[idx]);
}

// ------------------------------------------------------------- WMMA GEMM
// C = relu(A(MxK,bf16) * Bt(NxK,bf16)^T + bias)
// Block tile 64x128x32, 8 waves (2x4), wave tile 32x32 (2x2 WMMA accumulators),
// double-buffered LDS with async global->LDS staging.
#define BM 64
#define BN 128
#define BK 32
#define LDA 40   // padded LDS stride: 80B rows, conflict-free ds_load_b128

__global__ void __launch_bounds__(256)
gemm_bias_relu(const __bf16* __restrict__ A, const __bf16* __restrict__ Bt,
               const float* __restrict__ bias, int M, int N, int K,
               __bf16* __restrict__ outBf, float* __restrict__ outF) {
    __shared__ __attribute__((aligned(16))) __bf16 As[2][BM][LDA];
    __shared__ __attribute__((aligned(16))) __bf16 Bs[2][BN][LDA];

    const int tid  = threadIdx.x;
    const int wave = tid >> 5;
    const int lane = tid & 31;
    const int bm = blockIdx.x * BM;
    const int bn = blockIdx.y * BN;
    const int wm = wave >> 2;            // 0..1 -> row block wm*32
    const int wn = wave & 3;             // 0..3 -> col block wn*32
    const int lm   = lane & 15;
    const int half = (lane < 16) ? 0 : 8;   // ISA 16-bit A/B K-half per lane group

    const int arow = tid >> 2;           // 0..63
    const int acol = (tid & 3) * 8;      // 0,8,16,24

    const __bf16* gA  = A  + (long long)(bm + arow)       * K + acol;
    const __bf16* gB0 = Bt + (long long)(bn + arow)       * K + acol;
    const __bf16* gB1 = Bt + (long long)(bn + 64 + arow)  * K + acol;

    v8f acc00 = {}, acc01 = {}, acc10 = {}, acc11 = {};

    auto stage = [&](int b, int k0) {
        async_copy_b128(&As[b][arow][acol],      gA  + k0);
        async_copy_b128(&Bs[b][arow][acol],      gB0 + k0);
        async_copy_b128(&Bs[b][64 + arow][acol], gB1 + k0);
    };

    // prologue: stage first K slab into buffer 0
    stage(0, 0);
    wait_async_copies();
    __syncthreads();

    int buf = 0;
    for (int k0 = 0; k0 < K; k0 += BK) {
        int nbuf = buf ^ 1;
        if (k0 + BK < K) {
            stage(nbuf, k0 + BK);
            __builtin_prefetch(gA + k0 + 2 * BK, 0, 1);
        }

        union FragU { v16bf v; U128 q[2]; };
        FragU fa0, fa1, fb0, fb1;
        fa0.q[0] = *(const U128*)&As[buf][wm * 32      + lm][half];
        fa0.q[1] = *(const U128*)&As[buf][wm * 32      + lm][half + 16];
        fa1.q[0] = *(const U128*)&As[buf][wm * 32 + 16 + lm][half];
        fa1.q[1] = *(const U128*)&As[buf][wm * 32 + 16 + lm][half + 16];
        fb0.q[0] = *(const U128*)&Bs[buf][wn * 32      + lm][half];
        fb0.q[1] = *(const U128*)&Bs[buf][wn * 32      + lm][half + 16];
        fb1.q[0] = *(const U128*)&Bs[buf][wn * 32 + 16 + lm][half];
        fb1.q[1] = *(const U128*)&Bs[buf][wn * 32 + 16 + lm][half + 16];

        acc00 = __builtin_amdgcn_wmma_f32_16x16x32_bf16(false, fa0.v, false, fb0.v,
                                                        (short)0, acc00, false, false);
        acc01 = __builtin_amdgcn_wmma_f32_16x16x32_bf16(false, fa0.v, false, fb1.v,
                                                        (short)0, acc01, false, false);
        acc10 = __builtin_amdgcn_wmma_f32_16x16x32_bf16(false, fa1.v, false, fb0.v,
                                                        (short)0, acc10, false, false);
        acc11 = __builtin_amdgcn_wmma_f32_16x16x32_bf16(false, fa1.v, false, fb1.v,
                                                        (short)0, acc11, false, false);

        wait_async_copies();   // next slab landed (no-op on last iteration)
        __syncthreads();
        buf = nbuf;
    }

    // Epilogue per ISA C/D layout: VGPR e -> M = e (lanes 0-15) / e+8 (lanes 16-31)
    const int gn0 = bn + wn * 32 + lm;
    const int gn1 = gn0 + 16;
    const float bv0 = bias[gn0];
    const float bv1 = bias[gn1];
    #pragma unroll
    for (int e = 0; e < 8; ++e) {
        int rowoff = (lane < 16) ? e : (e + 8);
        int gm0 = bm + wm * 32 + rowoff;
        int gm1 = gm0 + 16;
        float v00 = fmaxf(acc00[e] + bv0, 0.f);
        float v01 = fmaxf(acc01[e] + bv1, 0.f);
        float v10 = fmaxf(acc10[e] + bv0, 0.f);
        float v11 = fmaxf(acc11[e] + bv1, 0.f);
        if (outBf) {
            outBf[(long long)gm0 * N + gn0] = f2bf(v00);
            outBf[(long long)gm0 * N + gn1] = f2bf(v01);
            outBf[(long long)gm1 * N + gn0] = f2bf(v10);
            outBf[(long long)gm1 * N + gn1] = f2bf(v11);
        }
        if (outF) {
            outF[(long long)gm0 * N + gn0] = v00;
            outF[(long long)gm0 * N + gn1] = v01;
            outF[(long long)gm1 * N + gn0] = v10;
            outF[(long long)gm1 * N + gn1] = v11;
        }
    }
}

// --------------------------------------------------------------- heads
__global__ void __launch_bounds__(256)
head_kernel(const float* __restrict__ H,
            const float* __restrict__ Wc, const float* __restrict__ bc,
            const float* __restrict__ Wo, const float* __restrict__ bo,
            const float* __restrict__ Wa, const float* __restrict__ ba,
            const float* __restrict__ bevAbs,
            float* __restrict__ objsoft, float* __restrict__ boxes,
            float* __restrict__ orient, float* __restrict__ scores) {
    __shared__ float red[256 * 8];
    int b = blockIdx.x, tid = threadIdx.x;
    float acc[8] = {0, 0, 0, 0, 0, 0, 0, 0};
    const float* h = H + (long long)b * HD;
    int kb = tid * 8;
    #pragma unroll
    for (int kk = 0; kk < 8; ++kk) {
        int k = kb + kk;
        float hv = h[k];
        acc[0] += hv * Wc[k * 2 + 0];
        acc[1] += hv * Wc[k * 2 + 1];
        acc[2] += hv * Wo[k * 10 + 0];
        acc[3] += hv * Wo[k * 10 + 1];
        acc[4] += hv * Wo[k * 10 + 2];
        acc[5] += hv * Wo[k * 10 + 3];
        acc[6] += hv * Wa[k * 2 + 0];
        acc[7] += hv * Wa[k * 2 + 1];
    }
    #pragma unroll
    for (int j = 0; j < 8; ++j) red[tid * 8 + j] = acc[j];
    __syncthreads();
    for (int s = 128; s > 0; s >>= 1) {
        if (tid < s) {
            #pragma unroll
            for (int j = 0; j < 8; ++j) red[tid * 8 + j] += red[(tid + s) * 8 + j];
        }
        __syncthreads();
    }
    if (tid == 0) {
        float o0 = red[0] + bc[0], o1 = red[1] + bc[1];
        float m = fmaxf(o0, o1);
        float e0 = expf(o0 - m), e1 = expf(o1 - m);
        float inv = 1.f / (e0 + e1);
        objsoft[b * 2 + 0] = e0 * inv;
        objsoft[b * 2 + 1] = e1 * inv;
        scores[b] = o1;                      // max over obj[:,1:] == obj[:,1]
        float pb[4];
        #pragma unroll
        for (int j = 0; j < 4; ++j) pb[j] = bevAbs[b * 4 + j] + 0.1f * (red[2 + j] + bo[j]);
        boxes[b * 4 + 0] = fminf(pb[0], pb[2]);
        boxes[b * 4 + 1] = fminf(pb[1], pb[3]);
        boxes[b * 4 + 2] = fmaxf(pb[0], pb[2]);
        boxes[b * 4 + 3] = fmaxf(pb[1], pb[3]);
        orient[b] = atan2f(red[7] + ba[1], red[6] + ba[0]);
    }
}

// ----------------------------------------------------------------- NMS
__global__ void __launch_bounds__(1024)
nms_kernel(const float* __restrict__ scores, const float* __restrict__ boxes,
           const float* __restrict__ objsoft, const float* __restrict__ orient,
           float* __restrict__ out) {
    __shared__ float sc[NBOX];
    __shared__ int   sid[NBOX];
    __shared__ float bx0[NBOX], bx1[NBOX], bx2[NBOX], bx3[NBOX], ar[NBOX];
    __shared__ int   keep[NBOX];
    __shared__ int   pre[NBOX];
    __shared__ int   sel[NMS_OUT];
    int tid = threadIdx.x;

    sc[tid]  = scores[tid];
    sid[tid] = tid;
    __syncthreads();

    // Bitonic sort: descending score, ascending idx tie-break (stable argsort(-s))
    for (int k = 2; k <= NBOX; k <<= 1) {
        for (int j = k >> 1; j > 0; j >>= 1) {
            int ixj = tid ^ j;
            if (ixj > tid) {
                float s1 = sc[tid], s2 = sc[ixj];
                int i1 = sid[tid], i2 = sid[ixj];
                bool desc = ((tid & k) == 0);
                bool sw = desc ? (s1 < s2 || (s1 == s2 && i1 > i2))
                               : (s1 > s2 || (s1 == s2 && i1 < i2));
                if (sw) { sc[tid] = s2; sc[ixj] = s1; sid[tid] = i2; sid[ixj] = i1; }
            }
            __syncthreads();
        }
    }

    int row = sid[tid];
    float x0 = boxes[row * 4 + 0], y0 = boxes[row * 4 + 1];
    float x1 = boxes[row * 4 + 2], y1 = boxes[row * 4 + 3];
    bx0[tid] = x0; bx1[tid] = y0; bx2[tid] = x1; bx3[tid] = y1;
    ar[tid] = fmaxf(x1 - x0, 0.f) * fmaxf(y1 - y0, 0.f);
    keep[tid] = 1;
    __syncthreads();

    for (int i = 0; i < NBOX - 1; ++i) {
        if (keep[i] && tid > i) {
            float lt0 = fmaxf(bx0[i], bx0[tid]), lt1 = fmaxf(bx1[i], bx1[tid]);
            float rb0 = fminf(bx2[i], bx2[tid]), rb1 = fminf(bx3[i], bx3[tid]);
            float w = fmaxf(rb0 - lt0, 0.f), h = fmaxf(rb1 - lt1, 0.f);
            float inter = w * h;
            float iou = inter / (ar[i] + ar[tid] - inter + 1e-8f);
            if (iou > NMS_THRV) keep[tid] = 0;
        }
        __syncthreads();
    }

    // inclusive prefix sum of keep -> reproduce argsort(rank)[:100]
    pre[tid] = keep[tid];
    __syncthreads();
    for (int s = 1; s < NBOX; s <<= 1) {
        int v = pre[tid] + ((tid >= s) ? pre[tid - s] : 0);
        __syncthreads();
        pre[tid] = v;
        __syncthreads();
    }
    int ktot = pre[NBOX - 1];
    int excl = pre[tid] - keep[tid];
    int slot = keep[tid] ? excl : (ktot + (tid - excl));
    if (slot < NMS_OUT) sel[slot] = row;
    __syncthreads();

    if (tid < NMS_OUT) {
        int r = sel[tid];
        out[tid * 7 + 0] = objsoft[r * 2 + 0];
        out[tid * 7 + 1] = objsoft[r * 2 + 1];
        out[tid * 7 + 2] = boxes[r * 4 + 0];
        out[tid * 7 + 3] = boxes[r * 4 + 1];
        out[tid * 7 + 4] = boxes[r * 4 + 2];
        out[tid * 7 + 5] = boxes[r * 4 + 3];
        out[tid * 7 + 6] = orient[r];
    }
}

// ------------------------------------------------------------- launcher
extern "C" void kernel_launch(void* const* d_in, const int* in_sizes, int n_in,
                              void* d_out, int out_size, void* d_ws, size_t ws_size,
                              hipStream_t stream) {
    const float* imgF   = (const float*)d_in[0];
    const float* bevF   = (const float*)d_in[1];
    const float* imgB   = (const float*)d_in[2];
    const float* bevB   = (const float*)d_in[3];
    const float* bevAbs = (const float*)d_in[4];
    const float* imgM   = (const float*)d_in[5];
    const float* bevM   = (const float*)d_in[6];
    const float* W1 = (const float*)d_in[7];
    const float* b1 = (const float*)d_in[8];
    const float* W2 = (const float*)d_in[9];
    const float* b2 = (const float*)d_in[10];
    const float* Wc = (const float*)d_in[11];
    const float* bc = (const float*)d_in[12];
    const float* Wo = (const float*)d_in[13];
    const float* bo = (const float*)d_in[14];
    const float* Wa = (const float*)d_in[15];
    const float* ba = (const float*)d_in[16];

    char* ws = (char*)d_ws;
    size_t off = 0;
    auto carve = [&](size_t bytes) -> char* {
        char* p = ws + off;
        off = (off + bytes + 255) & ~(size_t)255;
        return p;
    };
    __bf16* Xbf  = (__bf16*)carve((size_t)NBOX * D_IN * 2);
    __bf16* W1t  = (__bf16*)carve((size_t)D_IN * HD * 2);
    __bf16* H1bf = (__bf16*)carve((size_t)NBOX * HD * 2);
    __bf16* W2t  = (__bf16*)carve((size_t)HD * HD * 2);
    float*  H2f  = (float*)carve((size_t)NBOX * HD * 4);
    float*  objsoft  = (float*)carve((size_t)NBOX * 2 * 4);
    float*  boxesWs  = (float*)carve((size_t)NBOX * 4 * 4);
    float*  orientWs = (float*)carve((size_t)NBOX * 4);
    float*  scoresWs = (float*)carve((size_t)NBOX * 4);

    // 1. ROI align + fuse -> X bf16 (1024 x 1568)
    roi_fuse_kernel<<<NBOX, 256, 0, stream>>>(imgF, bevF, imgB, bevB, imgM, bevM, Xbf);

    // 2. Weight transpose+convert to bf16 (N x K layouts for WMMA B staging)
    {
        long long t1 = (long long)D_IN * HD;
        wconv_kernel<<<(unsigned)((t1 + 255) / 256), 256, 0, stream>>>(W1, W1t, D_IN, HD);
        long long t2 = (long long)HD * HD;
        wconv_kernel<<<(unsigned)((t2 + 255) / 256), 256, 0, stream>>>(W2, W2t, HD, HD);
    }

    // 3. GEMM1: H1 = relu(X @ W1 + b1), bf16 out
    {
        dim3 grid(NBOX / BM, HD / BN);
        gemm_bias_relu<<<grid, 256, 0, stream>>>(Xbf, W1t, b1, NBOX, HD, D_IN,
                                                 H1bf, (float*)nullptr);
    }
    // 4. GEMM2: H2 = relu(H1 @ W2 + b2), f32 out
    {
        dim3 grid(NBOX / BM, HD / BN);
        gemm_bias_relu<<<grid, 256, 0, stream>>>(H1bf, W2t, b2, NBOX, HD, HD,
                                                 (__bf16*)nullptr, H2f);
    }

    // 5. Heads: obj softmax, decoded boxes, orientation, scores
    head_kernel<<<NBOX, 256, 0, stream>>>(H2f, Wc, bc, Wo, bo, Wa, ba, bevAbs,
                                          objsoft, boxesWs, orientWs, scoresWs);

    // 6. NMS + top-100 gather -> 100 x 7 output
    nms_kernel<<<1, 1024, 0, stream>>>(scoresWs, boxesWs, objsoft, orientWs,
                                       (float*)d_out);
}
